// TreeEncoder_TreeLSTM_dgl_82935818486045
// MI455X (gfx1250) — compile-verified, matrix operand order checked by reading, not run on hardware
//
#include <hip/hip_runtime.h>
#include <hip/hip_bf16.h>

// ---------------------------------------------------------------------------
// TreeLSTM (complete binary tree, bottom-up) for MI455X / gfx1250.
// B=128 trees, DEPTH=10 (N=1023 nodes), H=256, NINP=256.
// Per tree level: fused bf16-WMMA GEMM (Uf + U_iou packed as [512 x 1280])
// + gate nonlinearities. h stored bf16 (next level's GEMM input), c stored
// f32 (recurrent accumulation). Each workgroup: 32 rows staged in LDS,
// 4 waves x (2 M-tiles x 5 N-tiles) WMMA accumulation, gates in-register.
// ---------------------------------------------------------------------------

typedef __bf16 bf16_t;
typedef __attribute__((ext_vector_type(16))) __bf16 v16bf;
typedef __attribute__((ext_vector_type(8)))  __bf16 v8bf;
typedef __attribute__((ext_vector_type(8)))  float  v8f;

#define TB 128           // batch (trees)
#define TH 256           // hidden
#define TNINP 256
#define TN 1023
#define LEAF_BASE 511    // 2^9 - 1
#define NLEAF 512
#define MROWS 32         // rows per workgroup (2 M-tiles of 16)

// packed weight geometry
#define WCAT_KT 16       // 512/32 k tiles  ([Uf(512) | U_iou(768)] cols)
#define WIOU_KT 8        // 256/32 k tiles  (W_iou, 768 cols)

#define WCAT_ELEMS (512 * 1280)
#define WIOU_ELEMS (256 * 768)

// workspace layout (bytes)
#define OFF_WCAT  0u
#define OFF_WIOU  1310720u                       // 512*1280*2
#define OFF_BCAT  (OFF_WIOU + 393216u)           // 256*768*2
#define OFF_HA    (OFF_BCAT + 8192u)
#define OFF_CA    (OFF_HA + 33554432u)           // 512*128*256 bf16
#define OFF_HB    (OFF_CA + 67108864u)           // 512*128*256 f32
#define OFF_CB    (OFF_HB + 33554432u)

#define LDS_STRIDE_L 264  // leaf: 256 + 8 pad (16B aligned, bank staggered)
#define LDS_STRIDE_I 520  // internal: 512 + 8 pad

__device__ __forceinline__ float dev_sigmoid(float x) {
    return 1.0f / (1.0f + __expf(-x));
}

// A-fragment (16x32 bf16) from a row-major LDS tile with padded stride.
// Per ISA 7.12.2: lanes 0-15 hold M=lane, K = k0+{0..7, 16..23};
//                 lanes 16-31 hold M=lane-16, K = k0+{8..15, 24..31}.
__device__ __forceinline__ v16bf load_a_frag(const bf16_t* lds, int stride,
                                             int k0, int lane) {
    const int m = lane & 15, half = lane >> 4;
    const bf16_t* p = lds + m * stride + k0 + 8 * half;
    v8bf lo = *(const v8bf*)p;
    v8bf hi = *(const v8bf*)(p + 16);
    v16bf a;
#pragma unroll
    for (int i = 0; i < 8; ++i) { a[i] = lo[i]; a[8 + i] = hi[i]; }
    return a;
}

// B-fragment: pre-packed so each lane's 16 bf16 are contiguous (32B load).
__device__ __forceinline__ v16bf load_b_frag(const bf16_t* wp, int ct, int kt,
                                             int ktiles, int lane) {
    return *(const v16bf*)(wp + (((size_t)ct * ktiles + kt) * 32 + lane) * 16);
}

__device__ __forceinline__ v8f wmma_bf16(v16bf a, v16bf b, v8f c) {
    return __builtin_amdgcn_wmma_f32_16x16x32_bf16(
        /*neg_a=*/false, a, /*neg_b=*/false, b,
        /*c_mod=*/(short)0, c, /*reuse_a=*/false, /*reuse_b=*/false);
}

// ---------------------------------------------------------------------------
// Pack weights into bf16 WMMA B-fragment order + fused bias vector.
// wcat cols: [0,512) = Uf_W, [512,1280) = U_iou.
// bcat:      [0,512) = Uf_b, [512,1280) = b_iou.
// ---------------------------------------------------------------------------
__global__ void pack_weights_kernel(const float* __restrict__ W_iou,
                                    const float* __restrict__ U_iou,
                                    const float* __restrict__ b_iou,
                                    const float* __restrict__ Uf_W,
                                    const float* __restrict__ Uf_b,
                                    bf16_t* __restrict__ wcat,
                                    bf16_t* __restrict__ wiou,
                                    float* __restrict__ bcat) {
    int idx = blockIdx.x * 256 + threadIdx.x;
    if (idx < WCAT_ELEMS) {
        int k = idx / 1280, col = idx % 1280;
        float v = (col < 512) ? Uf_W[k * 512 + col] : U_iou[k * 768 + (col - 512)];
        int ct = col >> 4, nn = col & 15, kt = k >> 5, kk = k & 31;
        int half = (kk >> 3) & 1;
        int lane = nn + 16 * half;
        int slot = (kk & 7) + 8 * (kk >> 4);
        wcat[(((size_t)ct * WCAT_KT + kt) * 32 + lane) * 16 + slot] = (bf16_t)v;
    } else if (idx < WCAT_ELEMS + WIOU_ELEMS) {
        int e = idx - WCAT_ELEMS;
        int k = e / 768, col = e % 768;
        float v = W_iou[k * 768 + col];
        int ct = col >> 4, nn = col & 15, kt = k >> 5, kk = k & 31;
        int half = (kk >> 3) & 1;
        int lane = nn + 16 * half;
        int slot = (kk & 7) + 8 * (kk >> 4);
        wiou[(((size_t)ct * WIOU_KT + kt) * 32 + lane) * 16 + slot] = (bf16_t)v;
    } else if (idx < WCAT_ELEMS + WIOU_ELEMS + 1280) {
        int e = idx - WCAT_ELEMS - WIOU_ELEMS;
        bcat[e] = (e < 512) ? Uf_b[e] : b_iou[e - 512];
    }
}

// ---------------------------------------------------------------------------
// Leaf level: iou = (wemb[wordid*mask] @ W_iou) * mask + b_iou; gates, cprev=0
// grid: (B*512/32 row groups, 4 column groups), block = 128 (4 waves).
// ---------------------------------------------------------------------------
__global__ __launch_bounds__(128) void tree_leaf_kernel(
    const int* __restrict__ wordid, const int* __restrict__ maskp,
    const float* __restrict__ wemb,
    bf16_t* __restrict__ h_out, float* __restrict__ c_out,
    const bf16_t* __restrict__ wiou, const float* __restrict__ bcat) {
    const int tile = blockIdx.x, cg = blockIdx.y;
    const int t = threadIdx.x, lane = t & 31, wave = t >> 5;

    __shared__ __align__(16) bf16_t atile[MROWS * LDS_STRIDE_L];
    __shared__ float rmask[MROWS];

    // Stage A: gather 32 embedding rows, convert f32 -> bf16.
    // 4 threads per row, each covers 64 columns.
    {
        int row = t >> 2, quarter = t & 3;
        int r = tile * MROWS + row;
        int b = r >> 9, q = r & 511;
        int node = b * TN + LEAF_BASE + q;
        int mk = maskp[node];
        int w = wordid[node] * mk;
        if (quarter == 0) rmask[row] = (float)mk;
        const float4* src =
            (const float4*)(wemb + (size_t)w * TNINP + quarter * 64);
        bf16_t* dst = atile + row * LDS_STRIDE_L + quarter * 64;
#pragma unroll
        for (int i = 0; i < 16; ++i) {
            float4 f = src[i];
            dst[i * 4 + 0] = (bf16_t)f.x;
            dst[i * 4 + 1] = (bf16_t)f.y;
            dst[i * 4 + 2] = (bf16_t)f.z;
            dst[i * 4 + 3] = (bf16_t)f.w;
        }
    }
    __syncthreads();

    const int j0 = (cg * 4 + wave) * 16;  // H column of this wave
    const int ct = j0 >> 4;
    v8f acc_i[2] = {{}, {}}, acc_o[2] = {{}, {}}, acc_u[2] = {{}, {}};
#pragma unroll
    for (int kt = 0; kt < WIOU_KT; ++kt) {
        v16bf bi = load_b_frag(wiou, ct, kt, WIOU_KT, lane);
        v16bf bo = load_b_frag(wiou, ct + 16, kt, WIOU_KT, lane);
        v16bf bu = load_b_frag(wiou, ct + 32, kt, WIOU_KT, lane);
#pragma unroll
        for (int mt = 0; mt < 2; ++mt) {
            v16bf a = load_a_frag(atile + mt * 16 * LDS_STRIDE_L,
                                  LDS_STRIDE_L, kt * 32, lane);
            acc_i[mt] = wmma_bf16(a, bi, acc_i[mt]);
            acc_o[mt] = wmma_bf16(a, bo, acc_o[mt]);
            acc_u[mt] = wmma_bf16(a, bu, acc_u[mt]);
        }
    }

    // Gates (leaf: c_prev = 0).  D layout: lane = n + 16*(m>=8), vgpr = m%8.
    const int nidx = lane & 15, mhalf = lane >> 4;
    const int col = j0 + nidx;
    const float bias_i = bcat[512 + col];
    const float bias_o = bcat[768 + col];
    const float bias_u = bcat[1024 + col];
#pragma unroll
    for (int mt = 0; mt < 2; ++mt) {
#pragma unroll
        for (int v = 0; v < 8; ++v) {
            int m = mt * 16 + v + 8 * mhalf;
            int r = tile * MROWS + m;
            float mskv = rmask[m];
            float iv = acc_i[mt][v] * mskv + bias_i;
            float ov = acc_o[mt][v] * mskv + bias_o;
            float uv = acc_u[mt][v] * mskv + bias_u;
            float cnew = dev_sigmoid(iv) * tanhf(uv);
            float hnew = dev_sigmoid(ov) * tanhf(cnew);
            size_t orow = (size_t)r * TH + col;
            h_out[orow] = (bf16_t)hnew;
            c_out[orow] = cnew;
        }
    }
}

// ---------------------------------------------------------------------------
// Internal level: h_cat @ [Uf | U_iou] (fused 5-tile WMMA) + gates.
// grid: (B*n/32 row groups, 4 column groups), block = 128 (4 waves).
// ---------------------------------------------------------------------------
__global__ __launch_bounds__(128) void tree_level_kernel(
    const bf16_t* __restrict__ h_child, const float* __restrict__ c_child,
    bf16_t* __restrict__ h_out, float* __restrict__ c_out,
    float* __restrict__ h_root, float* __restrict__ c_root,
    const bf16_t* __restrict__ wcat, const float* __restrict__ bcat,
    int log2n) {
    const int n = 1 << log2n;
    const int tile = blockIdx.x, cg = blockIdx.y;
    const int t = threadIdx.x, lane = t & 31, wave = t >> 5;

    __shared__ __align__(16) bf16_t atile[MROWS * LDS_STRIDE_I];

    // Stage h_cat: cols [0,256) = left child h, [256,512) = right child h.
    // 4 threads per row; quarter selects (side, 128-col half of that side).
    {
        int row = t >> 2, quarter = t & 3;
        int side = quarter >> 1, off = (quarter & 1) * 128;
        int r = tile * MROWS + row;
        int b = r >> log2n, q = r & (n - 1);
        size_t childrow = ((size_t)b << (log2n + 1)) + 2 * q + side;
        const uint4* src = (const uint4*)(h_child + childrow * TH + off);
        uint4* dst = (uint4*)(atile + row * LDS_STRIDE_I + side * 256 + off);
#pragma unroll
        for (int i = 0; i < 16; ++i) dst[i] = src[i];
    }
    __syncthreads();

    const int j0 = (cg * 4 + wave) * 16;
    const int ct = j0 >> 4;  // 0..15 inside each 256-wide block
    v8f acc_f0[2] = {{}, {}}, acc_f1[2] = {{}, {}};
    v8f acc_i[2] = {{}, {}}, acc_o[2] = {{}, {}}, acc_u[2] = {{}, {}};
    for (int kt = 0; kt < WCAT_KT; ++kt) {
        v16bf bf0 = load_b_frag(wcat, ct, kt, WCAT_KT, lane);       // Uf col j
        v16bf bf1 = load_b_frag(wcat, ct + 16, kt, WCAT_KT, lane);  // Uf 256+j
        v16bf bi  = load_b_frag(wcat, ct + 32, kt, WCAT_KT, lane);  // U_iou i
        v16bf bo  = load_b_frag(wcat, ct + 48, kt, WCAT_KT, lane);  // U_iou o
        v16bf bu  = load_b_frag(wcat, ct + 64, kt, WCAT_KT, lane);  // U_iou u
#pragma unroll
        for (int mt = 0; mt < 2; ++mt) {
            v16bf a = load_a_frag(atile + mt * 16 * LDS_STRIDE_I,
                                  LDS_STRIDE_I, kt * 32, lane);
            acc_f0[mt] = wmma_bf16(a, bf0, acc_f0[mt]);
            acc_f1[mt] = wmma_bf16(a, bf1, acc_f1[mt]);
            acc_i[mt]  = wmma_bf16(a, bi, acc_i[mt]);
            acc_o[mt]  = wmma_bf16(a, bo, acc_o[mt]);
            acc_u[mt]  = wmma_bf16(a, bu, acc_u[mt]);
        }
    }

    const int nidx = lane & 15, mhalf = lane >> 4;
    const int col = j0 + nidx;
    const float bias_f0 = bcat[col];
    const float bias_f1 = bcat[256 + col];
    const float bias_i = bcat[512 + col];
    const float bias_o = bcat[768 + col];
    const float bias_u = bcat[1024 + col];
#pragma unroll
    for (int mt = 0; mt < 2; ++mt) {
#pragma unroll
        for (int v = 0; v < 8; ++v) {
            int m = mt * 16 + v + 8 * mhalf;
            int r = tile * MROWS + m;
            int b = r >> log2n, q = r & (n - 1);
            size_t cl = ((size_t)b << (log2n + 1)) + 2 * q;
            float cleft = c_child[cl * TH + col];
            float cright = c_child[(cl + 1) * TH + col];
            float f0 = dev_sigmoid(acc_f0[mt][v] + bias_f0);
            float f1 = dev_sigmoid(acc_f1[mt][v] + bias_f1);
            float cprev = f0 * cleft + f1 * cright;
            float iv = acc_i[mt][v] + bias_i;
            float ov = acc_o[mt][v] + bias_o;
            float uv = acc_u[mt][v] + bias_u;
            float cnew = dev_sigmoid(iv) * tanhf(uv) + cprev;
            float hnew = dev_sigmoid(ov) * tanhf(cnew);
            size_t orow = (size_t)r * TH + col;
            h_out[orow] = (bf16_t)hnew;
            c_out[orow] = cnew;
            if (h_root != nullptr) {
                h_root[orow] = hnew;  // root: n==1 -> orow = b*H + col, [1,B,H]
                c_root[orow] = cnew;
            }
        }
    }
}

// ---------------------------------------------------------------------------
extern "C" void kernel_launch(void* const* d_in, const int* in_sizes, int n_in,
                              void* d_out, int out_size, void* d_ws, size_t ws_size,
                              hipStream_t stream) {
    (void)in_sizes; (void)n_in; (void)out_size; (void)ws_size;
    const int*   wordid = (const int*)d_in[0];
    const int*   maskp  = (const int*)d_in[1];
    const float* wemb   = (const float*)d_in[2];
    const float* W_iou  = (const float*)d_in[3];
    const float* U_iou  = (const float*)d_in[4];
    const float* b_iou  = (const float*)d_in[5];
    const float* Uf_W   = (const float*)d_in[6];
    const float* Uf_b   = (const float*)d_in[7];

    char* ws = (char*)d_ws;
    bf16_t* wcat = (bf16_t*)(ws + OFF_WCAT);
    bf16_t* wiou = (bf16_t*)(ws + OFF_WIOU);
    float*  bcat = (float*)(ws + OFF_BCAT);
    bf16_t* hA = (bf16_t*)(ws + OFF_HA);
    float*  cA = (float*)(ws + OFF_CA);
    bf16_t* hB = (bf16_t*)(ws + OFF_HB);
    float*  cB = (float*)(ws + OFF_CB);
    float* out = (float*)d_out;

    // 1) Pack weights to bf16 fragment order (small, once per call).
    {
        int total = WCAT_ELEMS + WIOU_ELEMS + 1280;
        pack_weights_kernel<<<(total + 255) / 256, 256, 0, stream>>>(
            W_iou, U_iou, b_iou, Uf_W, Uf_b, wcat, wiou, bcat);
    }

    // 2) Leaf level -> slot A.
    {
        dim3 grid(TB * NLEAF / MROWS, 4);
        tree_leaf_kernel<<<grid, 128, 0, stream>>>(wordid, maskp, wemb,
                                                   hA, cA, wiou, bcat);
    }

    // 3) Internal levels 8..0, ping-pong A/B; root writes f32 into d_out.
    bf16_t* hc = hA; float* cc = cA;
    bf16_t* hn = hB; float* cn = cB;
    for (int lvl = 8; lvl >= 0; --lvl) {
        int n = 1 << lvl;
        int rows = TB * n;
        dim3 grid(rows / MROWS, 4);
        float* hr = (lvl == 0) ? out : nullptr;
        float* cr = (lvl == 0) ? (out + TB * TH) : nullptr;
        tree_level_kernel<<<grid, 128, 0, stream>>>(hc, cc, hn, cn, hr, cr,
                                                    wcat, bcat, lvl);
        bf16_t* th = hc; hc = hn; hn = th;
        float* tc = cc; cc = cn; cn = tc;
    }
}
